// HSDTP_36764920054616
// MI455X (gfx1250) — matrix operand order repaired and checked by reading, not compiled, not run on hardware
//
#include <hip/hip_runtime.h>

typedef __attribute__((ext_vector_type(16))) __bf16 v16bf;
typedef __attribute__((ext_vector_type(8)))  float  v8f;
typedef int v4i_vs __attribute__((vector_size(16)));   // matches builtin param type

union FragAB {
    v16bf v;
    uint4 q[2];
};

#define AS1 __attribute__((address_space(1)))
#define AS3 __attribute__((address_space(3)))

#if __has_builtin(__builtin_amdgcn_global_load_async_to_lds_b128) && \
    __has_builtin(__builtin_amdgcn_s_wait_asynccnt)
#define USE_ASYNC_LDS 1
#else
#define USE_ASYNC_LDS 0
#endif

__device__ __forceinline__ unsigned short f32_to_bf16(float f) {
    unsigned int u = __float_as_uint(f);
    unsigned int r = 0x7FFFu + ((u >> 16) & 1u);   // round-to-nearest-even
    u += r;
    return (unsigned short)(u >> 16);
}

// Copy 16B global -> LDS. Async path uses the CDNA5 async-to-LDS engine
// (ASYNCcnt-tracked); fallback is global_load + ds_store.
__device__ __forceinline__ void copy16_g2l(const unsigned short* g,
                                           unsigned short* l) {
#if USE_ASYNC_LDS
    __builtin_amdgcn_global_load_async_to_lds_b128(
        (AS1 v4i_vs*)(unsigned long long)g,
        (AS3 v4i_vs*)(unsigned long long)l, 0, 0);
#else
    *(uint4*)l = *(const uint4*)g;
#endif
}

// ---------------------------------------------------------------------------
// Transpose + convert: X f32 [R rows = features, C cols = batch] (row-major)
//   -> Xt bf16 [C][R]  (batch-major, feature index contiguous)
// ---------------------------------------------------------------------------
__global__ __launch_bounds__(256) void xpose_cvt_kernel(
    const float* __restrict__ X, unsigned short* __restrict__ Xt, int R, int C)
{
    __shared__ unsigned short tile[32][33];
    const int c0 = blockIdx.x * 32;
    const int r0 = blockIdx.y * 32;
    const int tx = threadIdx.x;           // 0..31
    for (int i = threadIdx.y; i < 32; i += 8)
        tile[i][tx] = f32_to_bf16(X[(size_t)(r0 + i) * C + c0 + tx]);
    __syncthreads();
    for (int i = threadIdx.y; i < 32; i += 8)
        Xt[(size_t)(c0 + i) * R + r0 + tx] = tile[tx][i];
}

// ---------------------------------------------------------------------------
// Elementwise convert W f32 -> bf16 (packed pairs)
// ---------------------------------------------------------------------------
__global__ __launch_bounds__(256) void cvt_w_kernel(
    const float2* __restrict__ W2, unsigned int* __restrict__ Wb2, long npairs)
{
    long stride = (long)gridDim.x * 256;
    for (long i = (long)blockIdx.x * 256 + threadIdx.x; i < npairs; i += stride) {
        float2 f = W2[i];
        Wb2[i] = (unsigned int)f32_to_bf16(f.x) |
                 ((unsigned int)f32_to_bf16(f.y) << 16);
    }
}

// ---------------------------------------------------------------------------
// bf16 WMMA GEMM:  Ct[n][m] = sum_k A[m][k] * Bt[n][k]
//   A  = weights bf16, [M][K] row-major
//   Bt = activations bf16, [N][K] row-major (batch-major transposed X)
//   Ct = f32 output,  [N][M] row-major (batch-major)
// Block: 256 threads = 8 waves (2 M x 4 N); block tile 128(M) x 256(N);
// wave tile 64x64 = 4x4 WMMA tiles; BK=32; double-buffered LDS staging fed
// by async global->LDS (ASYNCcnt) when available.
// ---------------------------------------------------------------------------
#define BK      32
#define LDSTR   40    // row stride in bf16 elems: 80B = 20 banks, 16B aligned

__global__ __launch_bounds__(256) void gemm_bf16_kernel(
    const unsigned short* __restrict__ A,
    const unsigned short* __restrict__ Bt,
    float* __restrict__ Ct,
    int M, int N, int K)
{
    __shared__ unsigned short lsA[2][128 * LDSTR];   // 2 x 10240 B
    __shared__ unsigned short lsB[2][256 * LDSTR];   // 2 x 20480 B

    const int tid  = threadIdx.x;
    const int lane = tid & 31;
    const int hl   = lane >> 4;     // half of wave (0/1)
    const int l16  = lane & 15;
    const int wave = tid >> 5;      // 0..7
    const int wm   = wave & 1;      // wave M index (0..1)
    const int wn   = wave >> 1;     // wave N index (0..3)

    const int m0 = blockIdx.y * 128;
    const int n0 = blockIdx.x * 256;

    // staging assignment
    const int arow  = tid >> 1;          // 0..127
    const int ahalf = tid & 1;           // 16-elem half of a 32-elem row
    const unsigned short* gA = A  + (size_t)(m0 + arow) * K + ahalf * 16;
    const unsigned short* gB = Bt + (size_t)(n0 + tid) * K;      // tid = 0..255

    v8f zero = {};
    v8f acc[4][4];
#pragma unroll
    for (int i = 0; i < 4; ++i)
#pragma unroll
        for (int j = 0; j < 4; ++j) acc[i][j] = zero;

    const int NT = K / BK;

    // ---- issue stage 0 ----
    {
        copy16_g2l(gA, &lsA[0][arow * LDSTR + ahalf * 16]);
#pragma unroll
        for (int c = 0; c < 4; ++c)
            copy16_g2l(gB + c * 8, &lsB[0][tid * LDSTR + c * 8]);
    }

    for (int kt = 0; kt < NT; ++kt) {
        const int cur = kt & 1;
        if (kt + 1 < NT) {
            const int nxt = cur ^ 1;
            __syncthreads();   // everyone done reading buf[nxt] (iter kt-1)
            const int ko = (kt + 1) * BK;
            copy16_g2l(gA + ko, &lsA[nxt][arow * LDSTR + ahalf * 16]);
#pragma unroll
            for (int c = 0; c < 4; ++c)
                copy16_g2l(gB + ko + c * 8, &lsB[nxt][tid * LDSTR + c * 8]);
#if USE_ASYNC_LDS
            __builtin_amdgcn_s_wait_asynccnt(6);  // stage-kt loads landed
#endif
        } else {
#if USE_ASYNC_LDS
            __builtin_amdgcn_s_wait_asynccnt(0);
#endif
        }
        __syncthreads();       // all waves' stage-kt data visible

        // B fragments for this wave (reused across the 4 M sub-tiles)
        FragAB b[4];
#pragma unroll
        for (int j = 0; j < 4; ++j) {
            const int nl = wn * 64 + j * 16 + l16;
            const unsigned short* p = &lsB[cur][nl * LDSTR + 16 * hl];
            b[j].q[0] = *(const uint4*)(p);
            b[j].q[1] = *(const uint4*)(p + 8);
        }
#pragma unroll
        for (int i = 0; i < 4; ++i) {
            const int ml = wm * 64 + i * 16 + l16;
            const unsigned short* p = &lsA[cur][ml * LDSTR + 16 * hl];
            FragAB a;
            a.q[0] = *(const uint4*)(p);
            a.q[1] = *(const uint4*)(p + 8);
#pragma unroll
            for (int j = 0; j < 4; ++j)
                acc[i][j] = __builtin_amdgcn_wmma_f32_16x16x32_bf16(
                    false, a.v, false, b[j].v, (short)0, acc[i][j],
                    false, false);
        }
    }

    // D layout: VGPR r -> row m = r + 8*half, col n = l16. Rows contiguous in
    // batch-major Ct -> two float4 stores per tile per lane.
#pragma unroll
    for (int i = 0; i < 4; ++i) {
#pragma unroll
        for (int j = 0; j < 4; ++j) {
            const int n = n0 + wn * 64 + j * 16 + l16;
            const int m = m0 + wm * 64 + i * 16 + hl * 8;
            float* p = Ct + (size_t)n * M + m;
            *(float4*)(p)     = make_float4(acc[i][j][0], acc[i][j][1],
                                            acc[i][j][2], acc[i][j][3]);
            *(float4*)(p + 4) = make_float4(acc[i][j][4], acc[i][j][5],
                                            acc[i][j][6], acc[i][j][7]);
        }
    }
}

// ---------------------------------------------------------------------------
// Fused ReLU + exact kWTA per batch column.
//   Yt [N][M] f32 in/out (masked in place); writes bf16 next-layer input.
// Exact k-th largest of (relu(y) - thr) via 32-step binary search on the
// order-preserving unsigned mapping of float bits, keys cached in LDS.
// ---------------------------------------------------------------------------
__global__ __launch_bounds__(256) void kwta_kernel(
    float* __restrict__ Yt, const float* __restrict__ thr,
    unsigned short* __restrict__ Xnext, int M, int k, int do_relu)
{
    __shared__ unsigned int keys[8192];
    __shared__ int s_cnt;

    const int n   = blockIdx.x;
    const int tid = threadIdx.x;
    float* y = Yt + (size_t)n * M;

    for (int m = tid; m < M; m += 256) {
        float v = y[m];
        if (do_relu) v = fmaxf(v, 0.0f);
        float a = v - thr[m];
        unsigned int b = __float_as_uint(a);
        keys[m] = (b & 0x80000000u) ? ~b : (b | 0x80000000u);
    }
    __syncthreads();

    unsigned int lo = 0u, hi = 0xFFFFFFFFu;
    for (int it = 0; it < 32 && lo < hi; ++it) {
        unsigned int d   = hi - lo;
        unsigned int mid = lo + (d >> 1) + (d & 1u);   // upper mid, no overflow
        if (tid == 0) s_cnt = 0;
        __syncthreads();
        int c = 0;
        for (int m = tid; m < M; m += 256) c += (keys[m] >= mid) ? 1 : 0;
        atomicAdd(&s_cnt, c);
        __syncthreads();
        if (s_cnt >= k) lo = mid; else hi = mid - 1;
        __syncthreads();
    }
    // lo == ordered key of the k-th largest activation; mask = key >= lo
    for (int m = tid; m < M; m += 256) {
        float v = y[m];
        if (do_relu) v = fmaxf(v, 0.0f);
        float o = (keys[m] >= lo) ? v : 0.0f;
        y[m] = o;
        if (Xnext) Xnext[(size_t)n * M + m] = f32_to_bf16(o);
    }
}

// ---------------------------------------------------------------------------
// Feature-axis softmax: Yt [N][1024] batch-major -> out [1024][N] (ref layout)
// ---------------------------------------------------------------------------
__global__ __launch_bounds__(256) void softmax_kernel(
    const float* __restrict__ Yt, float* __restrict__ out, int M, int N)
{
    __shared__ float red[256];
    const int n   = blockIdx.x;
    const int tid = threadIdx.x;
    const float* y = Yt + (size_t)n * M;

    float mx = -3.402823466e+38f;
    for (int m = tid; m < M; m += 256) mx = fmaxf(mx, y[m]);
    red[tid] = mx;
    __syncthreads();
    for (int s = 128; s > 0; s >>= 1) {
        if (tid < s) red[tid] = fmaxf(red[tid], red[tid + s]);
        __syncthreads();
    }
    const float mxAll = red[0];
    __syncthreads();

    float sum = 0.0f;
    for (int m = tid; m < M; m += 256) sum += expf(y[m] - mxAll);
    red[tid] = sum;
    __syncthreads();
    for (int s = 128; s > 0; s >>= 1) {
        if (tid < s) red[tid] += red[tid + s];
        __syncthreads();
    }
    const float inv = 1.0f / red[0];

    for (int m = tid; m < M; m += 256)
        out[(size_t)m * N + n] = expf(y[m] - mxAll) * inv;
}

// ---------------------------------------------------------------------------
// Host side
// ---------------------------------------------------------------------------
extern "C" void kernel_launch(void* const* d_in, const int* in_sizes, int n_in,
                              void* d_out, int out_size, void* d_ws, size_t ws_size,
                              hipStream_t stream) {
    (void)in_sizes; (void)n_in; (void)out_size; (void)ws_size;

    const float* x = (const float*)d_in[0];
    const float* W[4]   = { (const float*)d_in[1], (const float*)d_in[3],
                            (const float*)d_in[5], (const float*)d_in[7] };
    const float* thr[4] = { (const float*)d_in[2], (const float*)d_in[4],
                            (const float*)d_in[6], (const float*)d_in[8] };

    const int dims[5] = { 4096, 8192, 8192, 4096, 1024 };
    const int batch = 4096;

    char* ws = (char*)d_ws;
    unsigned short* Wb = (unsigned short*)ws;                         // 128 MiB (bf16 weights, per layer)
    unsigned short* Xt = (unsigned short*)(ws + (size_t)134217728);   //  64 MiB (bf16 X^T, batch-major)
    float*          Yt = (float*)(ws + (size_t)134217728 + 67108864); // 128 MiB (f32 Y^T, batch-major)

    // x f32 [4096, batch] -> Xt bf16 [batch][4096]
    xpose_cvt_kernel<<<dim3(batch / 32, dims[0] / 32), dim3(32, 8), 0, stream>>>(
        x, Xt, dims[0], batch);

    for (int i = 0; i < 4; ++i) {
        const int in_d  = dims[i];
        const int out_d = dims[i + 1];
        const long npairs = (long)out_d * in_d / 2;

        cvt_w_kernel<<<8192, 256, 0, stream>>>(
            (const float2*)W[i], (unsigned int*)Wb, npairs);

        gemm_bf16_kernel<<<dim3(batch / 256, out_d / 128), 256, 0, stream>>>(
            Wb, Xt, Yt, out_d, batch, in_d);

        const int k = out_d * 5 / 100;   // replicates int(0.05 * out_d)
        kwta_kernel<<<batch, 256, 0, stream>>>(
            Yt, thr[i], (i < 3) ? Xt : (unsigned short*)nullptr, out_d, k,
            (i < 3) ? 1 : 0);
    }

    softmax_kernel<<<batch, 256, 0, stream>>>(Yt, (float*)d_out, dims[4], batch);
}
